// GraphSAGE_76751065579701
// MI455X (gfx1250) — matrix operand order repaired and checked by reading, not compiled, not run on hardware
//
#include <hip/hip_runtime.h>
#include <stdint.h>

typedef __attribute__((ext_vector_type(16))) __bf16 v16bf;
typedef __attribute__((ext_vector_type(8)))  __bf16 v8bf;
typedef __attribute__((ext_vector_type(4)))  __bf16 v4bf;
typedef __attribute__((ext_vector_type(8)))  float  v8f;
typedef __attribute__((ext_vector_type(4)))  unsigned int v4u;
typedef __attribute__((ext_vector_type(8)))  int v8i;
typedef __attribute__((ext_vector_type(4)))  int v4i;

#define LDA 40   // padded LDS stride (bf16 elems) for A tile rows
#define LDB 40   // padded LDS stride (bf16 elems) for B tile cols (64B data + 16B pad)

#if __has_builtin(__builtin_amdgcn_tensor_load_to_lds) && \
    __has_builtin(__builtin_amdgcn_s_wait_tensorcnt)
#define SAGE_USE_TDM 1
#else
#define SAGE_USE_TDM 0
#endif

// ---------------------------------------------------------------------------
// Utility kernels
// ---------------------------------------------------------------------------
__global__ void zero_f32x4_kernel(float4* __restrict__ p, long long n4) {
  long long i = (long long)blockIdx.x * blockDim.x + threadIdx.x;
  long long stride = (long long)gridDim.x * blockDim.x;
  float4 z = {0.0f, 0.0f, 0.0f, 0.0f};
  for (; i < n4; i += stride) p[i] = z;
}

__global__ void count_deg_kernel(const long long* __restrict__ dst,
                                 float* __restrict__ deg, int E) {
  int e = blockIdx.x * blockDim.x + threadIdx.x;
  if (e < E) {
    int t = (int)dst[e];
    __hip_atomic_fetch_add(&deg[t], 1.0f, __ATOMIC_RELAXED,
                           __HIP_MEMORY_SCOPE_AGENT);
  }
}

__global__ void inv_deg_kernel(float* __restrict__ deg, int N) {
  int i = blockIdx.x * blockDim.x + threadIdx.x;
  if (i < N) deg[i] = 1.0f / fmaxf(deg[i], 1.0f);
}

// W [K][C] f32 (row-major) -> Wt [C][K] bf16 (transposed, for TDM / vector copy)
__global__ void w_to_bf16t_kernel(const float* __restrict__ W,
                                  __bf16* __restrict__ Wt, int K, int C) {
  int idx = blockIdx.x * blockDim.x + threadIdx.x;
  if (idx < K * C) {
    int k = idx / C;
    int c = idx % C;
    Wt[(long long)c * K + k] = (__bf16)W[idx];
  }
}

// Edge-parallel scatter-add: thread = (edge, 4-feature chunk).
__global__ void scatter_add_kernel(const float* __restrict__ x,
                                   const long long* __restrict__ src,
                                   const long long* __restrict__ dst,
                                   float* __restrict__ agg,
                                   int E, int d, int q_shift /* log2(d/4) */) {
  long long idx = (long long)blockIdx.x * blockDim.x + threadIdx.x;
  long long total = (long long)E << q_shift;
  if (idx >= total) return;
  int e  = (int)(idx >> q_shift);
  int f4 = (int)(idx & ((1 << q_shift) - 1));
  int s = (int)src[e];
  int t = (int)dst[e];
  const float4 v = *(const float4*)(x + (long long)s * d + f4 * 4);
  float* o = agg + (long long)t * d + f4 * 4;
  __hip_atomic_fetch_add(o + 0, v.x, __ATOMIC_RELAXED, __HIP_MEMORY_SCOPE_AGENT);
  __hip_atomic_fetch_add(o + 1, v.y, __ATOMIC_RELAXED, __HIP_MEMORY_SCOPE_AGENT);
  __hip_atomic_fetch_add(o + 2, v.z, __ATOMIC_RELAXED, __HIP_MEMORY_SCOPE_AGENT);
  __hip_atomic_fetch_add(o + 3, v.w, __ATOMIC_RELAXED, __HIP_MEMORY_SCOPE_AGENT);
}

// ---------------------------------------------------------------------------
// Fragment helpers
// ---------------------------------------------------------------------------
__device__ __forceinline__ v16bf frag16(const __bf16* p0, const __bf16* p1) {
  v8bf lo = *(const v8bf*)p0;
  v8bf hi = *(const v8bf*)p1;
  return __builtin_shufflevector(lo, hi, 0, 1, 2, 3, 4, 5, 6, 7,
                                         8, 9, 10, 11, 12, 13, 14, 15);
}

__device__ __forceinline__ v4bf pack4(float4 v, float s) {
  v4bf r;
  r[0] = (__bf16)(v.x * s);
  r[1] = (__bf16)(v.y * s);
  r[2] = (__bf16)(v.z * s);
  r[3] = (__bf16)(v.w * s);
  return r;
}

#if SAGE_USE_TDM
// One TDM descriptor: load a 2D tile [tile_dim1=ncols rows] x [tile_dim0=32
// bf16] from transposed weights Wt[c][k] (row stride Kdim elems) into LDS at
// lds_off, inserting 16B pad after each 64B row => LDS stride = LDB(40) bf16.
__device__ __forceinline__ void tdm_load_b(const __bf16* gsrc, unsigned lds_off,
                                           int Kdim, int ncols) {
  unsigned long long ga = (unsigned long long)(uintptr_t)gsrc;
  v4u g0;
  g0[0] = 1u;                                    // count=1 (valid user D#)
  g0[1] = lds_off;                               // lds_addr
  g0[2] = (unsigned)(ga & 0xFFFFFFFFu);          // global_addr[31:0]
  g0[3] = (unsigned)((ga >> 32) & 0x1FFFFFFu)    // global_addr[56:32]
          | (2u << 30);                          // type=2 ("image")
  v8i g1;
  g1[0] = (int)((1u << 16)        // data_size=1 (2 bytes)
                | (1u << 20)      // pad_enable
                | (3u << 22)      // pad_interval: 16 DWORDs (64B)
                | (3u << 25));    // pad_amount:   4 DWORDs (16B)
  g1[1] = (int)(((unsigned)Kdim & 0xFFFFu) << 16);            // tensor_dim0 lo16
  g1[2] = (int)((((unsigned)Kdim >> 16) & 0xFFFFu)            // tensor_dim0 hi16
                | (((unsigned)ncols & 0xFFFFu) << 16));       // tensor_dim1 lo16
  g1[3] = (int)((((unsigned)ncols >> 16) & 0xFFFFu)           // tensor_dim1 hi16
                | (32u << 16));                               // tile_dim0 = 32
  g1[4] = (int)((unsigned)ncols & 0xFFFFu);                   // tile_dim1, tile_dim2=0
  g1[5] = (int)(unsigned)Kdim;                                // tensor_dim0_stride[31:0]
  g1[6] = 0;                                                  // stride hi / dim1_stride lo
  g1[7] = 0;
  v4i z4 = {0, 0, 0, 0};
  v8i z8 = {0, 0, 0, 0, 0, 0, 0, 0};
  // clang-23 / therock-10.0 form: (g0, g1, g2, g3, extra, cpol)
  __builtin_amdgcn_tensor_load_to_lds(g0, g1, z4, z4, z8, 0);
}
#endif

// ---------------------------------------------------------------------------
// Fused SAGE GEMM:  out = relu?( (agg * rowscale) @ Wl + xroot @ Wr + bias )
// Block: 256 threads = 8 waves (2x4). Block tile 64 x NCOLS, K-step 32.
// bf16 WMMA 16x16x32, f32 accumulate. B tiles come from transposed bf16
// weights via the Tensor Data Mover (one instruction per K-step).
// ---------------------------------------------------------------------------
template <int NCOLS>
__global__ __launch_bounds__(256)
void sage_gemm_kernel(const float* __restrict__ Aagg,    // [N,K] summed neighbors
                      const float* __restrict__ rscale,  // [N]   1/max(deg,1)
                      const float* __restrict__ Aroot,   // [N,K] root features
                      const __bf16* __restrict__ Wlt,    // [NCOLS,K] bf16 (W_l^T)
                      const __bf16* __restrict__ Wrt,    // [NCOLS,K] bf16 (W_r^T)
                      const float* __restrict__ bias,    // [NCOLS]
                      float* __restrict__ out,           // [N,NCOLS]
                      int Nrows, int Kdim, int do_relu) {
  constexpr int NSUB = NCOLS / 64;   // 16-col frags per wave

  __shared__ __bf16 Alds[64 * LDA];      // 64 rows x 32 K   (row-major)
  __shared__ __bf16 Blds[NCOLS * LDB];   // NCOLS cols x 32 K (col-major slices)

  const int tid  = threadIdx.x;
  const int lane = tid & 31;
  const int w    = tid >> 5;             // 0..7
  const int wm   = w & 1;                // wave row (2)
  const int wn   = w >> 1;               // wave col (4)
  const int row0 = blockIdx.x * 64;

  const int l  = lane & 15;
  const int hb = lane >> 4;

  // A staging: rows clamped; OOB rows only feed OOB output rows, which the
  // guarded epilogue never stores.
  const int r0  = tid >> 3;                          // 0..31
  const int r1  = r0 + 32;                           // 32..63
  const int k4  = tid & 7;                           // 8 float4 per 32-K row
  const int gr0 = min(row0 + r0, Nrows - 1);
  const int gr1 = min(row0 + r1, Nrows - 1);
  const float sc0 = rscale[gr0];
  const float sc1 = rscale[gr1];

#if SAGE_USE_TDM
  const unsigned blds_off = (unsigned)(uintptr_t)(void*)&Blds[0];
#endif

  v8f acc[2][NSUB] = {};

  for (int pass = 0; pass < 2; ++pass) {
    const float*  Ag = pass ? Aroot : Aagg;
    const __bf16* Wt = pass ? Wrt : Wlt;
    const float s0 = pass ? 1.0f : sc0;
    const float s1 = pass ? 1.0f : sc1;
    const float* arow0 = Ag + (long long)gr0 * Kdim;
    const float* arow1 = Ag + (long long)gr1 * Kdim;

    for (int k0 = 0; k0 < Kdim; k0 += 32) {
      // ---- Stage B tile: 32 K x NCOLS cols, LDS layout [c][k] stride LDB ----
#if SAGE_USE_TDM
      if (w == 0) {
        tdm_load_b(Wt + k0, blds_off, Kdim, NCOLS);
        __builtin_amdgcn_s_wait_tensorcnt(0);
      }
#else
#pragma unroll
      for (int i = 0; i < NCOLS / 64; ++i) {
        int idx = i * 256 + tid;       // over NCOLS*4 chunks of 8 bf16
        int c  = idx >> 2;
        int kc = idx & 3;
        *(v8bf*)&Blds[c * LDB + kc * 8] =
            *(const v8bf*)(Wt + (long long)c * Kdim + k0 + kc * 8);
      }
#endif
      // ---- Stage A tile 64x32: two unconditional b128 loads per thread ----
      float4 va0 = *(const float4*)(arow0 + k0 + k4 * 4);
      float4 va1 = *(const float4*)(arow1 + k0 + k4 * 4);
      *(v4bf*)&Alds[r0 * LDA + k4 * 4] = pack4(va0, s0);
      *(v4bf*)&Alds[r1 * LDA + k4 * 4] = pack4(va1, s1);

      __syncthreads();

      // A frags: lanes 0-15 -> row, K {0..7,16..23}; lanes 16-31 -> K {8..15,24..31}
      v16bf a[2];
#pragma unroll
      for (int m = 0; m < 2; ++m) {
        const __bf16* pa = &Alds[(wm * 32 + m * 16 + l) * LDA + hb * 8];
        a[m] = frag16(pa, pa + 16);
      }
      // B frags: lanes 0-15 -> col, K {0..7,16..23}; lanes 16-31 -> K {8..15,24..31}
#pragma unroll
      for (int n = 0; n < NSUB; ++n) {
        const __bf16* pb = &Blds[(wn * (NSUB * 16) + n * 16 + l) * LDB + hb * 8];
        v16bf b = frag16(pb, pb + 16);
#pragma unroll
        for (int m = 0; m < 2; ++m) {
          acc[m][n] = __builtin_amdgcn_wmma_f32_16x16x32_bf16(
              false, a[m], false, b, (short)0, acc[m][n], false, false);
        }
      }
      __syncthreads();
    }
  }

  // Epilogue: C/D layout -> VGPR j holds row (hb*8 + j), column l of subtile.
#pragma unroll
  for (int n = 0; n < NSUB; ++n) {
    int col = wn * (NSUB * 16) + n * 16 + l;
    float bia = bias[col];
#pragma unroll
    for (int m = 0; m < 2; ++m) {
#pragma unroll
      for (int j = 0; j < 8; ++j) {
        int row = row0 + wm * 32 + m * 16 + hb * 8 + j;
        if (row < Nrows) {
          float v = acc[m][n][j] + bia;
          if (do_relu) v = fmaxf(v, 0.0f);
          out[(long long)row * NCOLS + col] = v;
        }
      }
    }
  }
}

// ---------------------------------------------------------------------------
// Host orchestration
// ---------------------------------------------------------------------------
extern "C" void kernel_launch(void* const* d_in, const int* in_sizes, int n_in,
                              void* d_out, int out_size, void* d_ws, size_t ws_size,
                              hipStream_t stream) {
  const float*     x   = (const float*)d_in[0];
  const long long* ei  = (const long long*)d_in[1];
  const float*     Wl0 = (const float*)d_in[2];
  const float*     b0  = (const float*)d_in[3];
  const float*     Wr0 = (const float*)d_in[4];
  const float*     Wl1 = (const float*)d_in[5];
  const float*     b1  = (const float*)d_in[6];
  const float*     Wr1 = (const float*)d_in[7];
  const float*     Wl2 = (const float*)d_in[8];
  const float*     b2  = (const float*)d_in[9];
  const float*     Wr2 = (const float*)d_in[10];

  const int N = in_sizes[0] / 128;   // 50000
  const int E = in_sizes[1] / 2;     // 800000
  const long long* src = ei;
  const long long* dst = ei + E;

  float* ws  = (float*)d_ws;
  float* deg = ws;                                   // N floats (1/deg after inv)
  float* agg = ws + (size_t)((N + 255) & ~255);      // N*256 floats
  float* h   = agg + (size_t)N * 256;                // N*256 floats
  __bf16* wt   = (__bf16*)(h + (size_t)N * 256);     // transposed bf16 weights
  __bf16* wl0t = wt;                 // [256][128]
  __bf16* wr0t = wl0t + 32768;       // [256][128]
  __bf16* wl1t = wr0t + 32768;       // [256][256]
  __bf16* wr1t = wl1t + 65536;       // [256][256]
  __bf16* wl2t = wr1t + 65536;       // [128][256]
  __bf16* wr2t = wl2t + 32768;       // [128][256]

  const int T = 256;
  const unsigned gN   = (unsigned)((N + T - 1) / T);
  const unsigned gE   = (unsigned)((E + T - 1) / T);
  const unsigned gM   = (unsigned)((N + 63) / 64);
  const unsigned gE32 = (unsigned)(((long long)E * 32 + T - 1) / T);
  const unsigned gE64 = (unsigned)(((long long)E * 64 + T - 1) / T);

  // weight conversion (tiny) + degree (once)
  w_to_bf16t_kernel<<<128, T, 0, stream>>>(Wl0, wl0t, 128, 256);
  w_to_bf16t_kernel<<<128, T, 0, stream>>>(Wr0, wr0t, 128, 256);
  w_to_bf16t_kernel<<<256, T, 0, stream>>>(Wl1, wl1t, 256, 256);
  w_to_bf16t_kernel<<<256, T, 0, stream>>>(Wr1, wr1t, 256, 256);
  w_to_bf16t_kernel<<<128, T, 0, stream>>>(Wl2, wl2t, 256, 128);
  w_to_bf16t_kernel<<<128, T, 0, stream>>>(Wr2, wr2t, 256, 128);

  zero_f32x4_kernel<<<256, T, 0, stream>>>((float4*)deg, (long long)N / 4);
  count_deg_kernel<<<gE, T, 0, stream>>>(dst, deg, E);
  inv_deg_kernel<<<gN, T, 0, stream>>>(deg, N);

  // ---- layer 0 (d_in=128 -> 256, relu) ----
  zero_f32x4_kernel<<<2048, T, 0, stream>>>((float4*)agg, (long long)N * 32);
  scatter_add_kernel<<<gE32, T, 0, stream>>>(x, src, dst, agg, E, 128, 5);
  sage_gemm_kernel<256><<<gM, T, 0, stream>>>(agg, deg, x, wl0t, wr0t, b0, h, N, 128, 1);

  // ---- layer 1 (256 -> 256, relu); out aliases in: block owns its rows ----
  zero_f32x4_kernel<<<4096, T, 0, stream>>>((float4*)agg, (long long)N * 64);
  scatter_add_kernel<<<gE64, T, 0, stream>>>(h, src, dst, agg, E, 256, 6);
  sage_gemm_kernel<256><<<gM, T, 0, stream>>>(agg, deg, h, wl1t, wr1t, b1, h, N, 256, 1);

  // ---- layer 2 (256 -> 128, no relu) ----
  zero_f32x4_kernel<<<4096, T, 0, stream>>>((float4*)agg, (long long)N * 64);
  scatter_add_kernel<<<gE64, T, 0, stream>>>(h, src, dst, agg, E, 256, 6);
  sage_gemm_kernel<128><<<gM, T, 0, stream>>>(agg, deg, h, wl2t, wr2t, b2,
                                              (float*)d_out, N, 256, 0);
}
